// SparseUNet3D_83674552861286
// MI455X (gfx1250) — matrix operand order repaired and checked by reading, not compiled, not run on hardware
//
#include <hip/hip_runtime.h>
#include <hip/hip_bf16.h>

// ---------------------------------------------------------------------------
// CDNA5 (gfx1250) sparse U-Net forward, bf16 WMMA (v_wmma_f32_16x16x32_bf16).
// - Weights prepacked to bf16 in B-fragment order (2 x b128 per fragment).
// - Activations kept in bf16 between layers (2 x b128 per A fragment).
// - Kernel templated over (rulebook mode, A-alignment): no runtime mode
//   branches in the hot loop.
// - Wave-uniform ballot skip of all-empty gather taps.
// ---------------------------------------------------------------------------

typedef __attribute__((ext_vector_type(16))) __bf16        v16bf;
typedef __attribute__((ext_vector_type(8)))  float         v8f;
typedef __attribute__((ext_vector_type(4)))  unsigned int  v4u;

__device__ __forceinline__ unsigned short f2bf(float f) {
  unsigned int u = __float_as_uint(f);
  u += 0x7FFFu + ((u >> 16) & 1u);      // round-to-nearest-even
  return (unsigned short)(u >> 16);
}

union ABfrag { v16bf v; unsigned short u[16]; v4u q[2]; };

#define GM_NBR   0   // idx = nbr[row*K + k]
#define GM_DENSE 1   // idx = row (K must be 1)
#define GM_UP    2   // idx = (koff[row]==k) ? par[row] : -1

// Pack fp32 weight [Ktaps][Cin][Co] into bf16 WMMA-B fragments:
// Wp[(((kt*nC + c)*nT + t)*32 + lane)*16 + e]
//   = bf16(W[kt][c*32 + (lane<16?0:16) + e][t*16 + (lane&15)])
__global__ __launch_bounds__(256)
void pack_w(const float* __restrict__ W, unsigned short* __restrict__ Wp,
            int Ktaps, int Cin, int Co)
{
  const int nT = Co >> 4;
  const int nC = (Cin + 31) >> 5;
  size_t total = (size_t)Ktaps * nC * nT * 512;
  size_t i = (size_t)blockIdx.x * 256 + threadIdx.x;
  if (i >= total) return;
  int e = (int)(i & 15);
  int l = (int)((i >> 4) & 31);
  size_t r = i >> 9;
  int t  = (int)(r % nT); r /= nT;
  int c  = (int)(r % nC);
  int kt = (int)(r / nC);
  int K   = c * 32 + ((l < 16) ? 0 : 16) + e;
  int col = t * 16 + (l & 15);
  float v = (K < Cin && col < Co) ? W[((size_t)kt * Cin + K) * Co + col] : 0.0f;
  Wp[i] = f2bf(v);
}

__global__ __launch_bounds__(256)
void f32_to_bf16(const float* __restrict__ X, unsigned short* __restrict__ Y, size_t n)
{
  size_t i = (size_t)blockIdx.x * 256 + threadIdx.x;
  if (i < n) Y[i] = f2bf(X[i]);
}

// Generic rulebook gather-GEMM:  Y[n,:] (+)= sum_k X[idx(n,k),:] @ W[k]
// Block = 128 threads = 4 waves. Wave -> 32 rows x 64 cols (2x4 16x16 tiles).
template <int MODE, bool ALIGNED>
__device__ __forceinline__
void gconv_body(const unsigned short* __restrict__ X,
                const unsigned short* __restrict__ Wp,
                const int* __restrict__ nbr, const int* __restrict__ par,
                const int* __restrict__ koff, const float* __restrict__ bias,
                float* __restrict__ Y, unsigned short* __restrict__ Ybf,
                int N, int Cin, int Co, int Ktaps, int accum)
{
  const int lane    = threadIdx.x & 31;
  const int wave    = threadIdx.x >> 5;
  const int rowBase = (blockIdx.x * 4 + wave) * 32;
  if (rowBase >= N) return;                       // wave-uniform: EXEC stays full
  const int colBase = blockIdx.y * 64;
  const int nC = (Cin + 31) >> 5;
  const int nT = Co >> 4;                         // Co is always a multiple of 16

  v8f acc[2][4] = {};

  int  mrow[2], crow[2];
  bool rowOK[2];
#pragma unroll
  for (int m = 0; m < 2; ++m) {
    mrow[m]  = rowBase + m * 16 + (lane & 15);    // A-matrix: M = lane&15
    rowOK[m] = (mrow[m] < N);
    crow[m]  = rowOK[m] ? mrow[m] : (N - 1);      // clamped: loads always legal
  }

  int pidx[2] = {0, 0}, pk[2] = {-1, -1};
  if (MODE == GM_UP) {
#pragma unroll
    for (int m = 0; m < 2; ++m) {
      int p = par[crow[m]];
      int q = koff[crow[m]];
      pidx[m] = p;
      pk[m]   = rowOK[m] ? q : -1;
    }
  }

  // A k order used here (same as pack_w B): lane<16 -> K in {0..15}; lane>=16 -> {16..31}
  const int aK0 = (lane < 16) ? 0 : 16;
  const int aK1 = (lane < 16) ? 8 : 24;

  for (int kt = 0; kt < Ktaps; ++kt) {
    int gidx[2];
#pragma unroll
    for (int m = 0; m < 2; ++m) {
      if (MODE == GM_NBR) {
        int raw = nbr[(size_t)crow[m] * Ktaps + kt];   // unconditional (clamped row)
        gidx[m] = rowOK[m] ? raw : -1;
      } else if (MODE == GM_DENSE) {
        gidx[m] = crow[m];                             // always valid
      } else {
        gidx[m] = (pk[m] == kt) ? pidx[m] : -1;
      }
    }

    // Sparse rulebooks: skip taps with no present neighbor in this 32-row tile.
    if (MODE != GM_DENSE) {
      if (!__any(gidx[0] >= 0 || gidx[1] >= 0)) continue;   // wave-uniform
    }

    for (int cIdx = 0; cIdx < nC; ++cIdx) {
      const int c0 = cIdx << 5;
      ABfrag a[2];
#pragma unroll
      for (int m = 0; m < 2; ++m) {
        if (ALIGNED) {
          if (MODE == GM_DENSE) {
            const unsigned short* xr = X + (size_t)gidx[m] * Cin + c0;
            a[m].q[0] = *(const v4u*)(xr + aK0);
            a[m].q[1] = *(const v4u*)(xr + aK1);
          } else {
            v4u z = {0u, 0u, 0u, 0u};
            a[m].q[0] = z; a[m].q[1] = z;
            if (gidx[m] >= 0) {
              const unsigned short* xr = X + (size_t)gidx[m] * Cin + c0;
              a[m].q[0] = *(const v4u*)(xr + aK0);
              a[m].q[1] = *(const v4u*)(xr + aK1);
            }
          }
        } else {
          const unsigned short* xr = X + (size_t)(gidx[m] < 0 ? 0 : gidx[m]) * Cin;
#pragma unroll
          for (int e = 0; e < 8; ++e) {
            int ch0 = c0 + aK0 + e, ch1 = c0 + aK1 + e;
            a[m].u[e]     = (gidx[m] >= 0 && ch0 < Cin) ? xr[ch0] : (unsigned short)0;
            a[m].u[8 + e] = (gidx[m] >= 0 && ch1 < Cin) ? xr[ch1] : (unsigned short)0;
          }
        }
      }

      const unsigned short* wrow =
          Wp + (((size_t)kt * nC + cIdx) * nT + (colBase >> 4)) * 512 + (size_t)lane * 16;
#pragma unroll
      for (int nt = 0; nt < 4; ++nt) {
        if (colBase + nt * 16 < Co) {             // uniform guard
          ABfrag b;
          const unsigned short* bp = wrow + (size_t)nt * 512;
          b.q[0] = *(const v4u*)(bp);
          b.q[1] = *(const v4u*)(bp + 8);
          acc[0][nt] = __builtin_amdgcn_wmma_f32_16x16x32_bf16(
              false, a[0].v, false, b.v, (short)0, acc[0][nt], false, false);
          acc[1][nt] = __builtin_amdgcn_wmma_f32_16x16x32_bf16(
              false, a[1].v, false, b.v, (short)0, acc[1][nt], false, false);
        }
      }
    }
  }

  // C/D 16x16 layout: col = lane&15, row = vgpr_e + (lane<16 ? 0 : 8)
  const int rOff = (lane < 16) ? 0 : 8;
#pragma unroll
  for (int m = 0; m < 2; ++m) {
#pragma unroll
    for (int nt = 0; nt < 4; ++nt) {
      if (colBase + nt * 16 < Co) {
        const int col = colBase + nt * 16 + (lane & 15);
        const float bv = bias ? bias[col] : 0.0f;
#pragma unroll
        for (int e = 0; e < 8; ++e) {
          int r = rowBase + m * 16 + rOff + e;
          if (r < N) {
            size_t o = (size_t)r * Co + col;
            float v = acc[m][nt][e] + bv;
            if (accum) v += Y[o];
            Y[o] = v;
            if (Ybf) Ybf[o] = f2bf(v);
          }
        }
      }
    }
  }
}

#define GCONV_KERNEL(name, MODE, ALIGNED)                                        \
  __global__ __launch_bounds__(128)                                              \
  void name(const unsigned short* __restrict__ X,                                \
            const unsigned short* __restrict__ Wp,                               \
            const int* __restrict__ nbr, const int* __restrict__ par,            \
            const int* __restrict__ koff, const float* __restrict__ bias,        \
            float* __restrict__ Y, unsigned short* __restrict__ Ybf,             \
            int N, int Cin, int Co, int Ktaps, int accum) {                      \
    gconv_body<MODE, ALIGNED>(X, Wp, nbr, par, koff, bias, Y, Ybf,               \
                              N, Cin, Co, Ktaps, accum);                         \
  }

GCONV_KERNEL(gconv_nbr_a, GM_NBR, true)     // 27/8-tap rulebook, Cin % 32 == 0
GCONV_KERNEL(gconv_nbr_u, GM_NBR, false)    // stem (Cin = 5)
GCONV_KERNEL(gconv_dense, GM_DENSE, true)   // 1x1 GEMMs
GCONV_KERNEL(gconv_up,    GM_UP,    true)   // inverse conv (parent/koff)

// Deterministic BN reduce: one block per channel, tree reduction (no atomics).
__global__ __launch_bounds__(256)
void bn_reduce(const float* __restrict__ X, float* __restrict__ stats, int N, int C)
{
  const int c = blockIdx.x;
  float s = 0.f, ss = 0.f;
  for (int r = threadIdx.x; r < N; r += 256) {
    float v = X[(size_t)r * C + c];
    s += v; ss += v * v;
  }
  __shared__ float sh[512];
  sh[threadIdx.x] = s; sh[256 + threadIdx.x] = ss;
  __syncthreads();
  for (int st = 128; st > 0; st >>= 1) {
    if (threadIdx.x < st) {
      sh[threadIdx.x]       += sh[threadIdx.x + st];
      sh[256 + threadIdx.x] += sh[256 + threadIdx.x + st];
    }
    __syncthreads();
  }
  if (threadIdx.x == 0) { stats[c] = sh[0]; stats[C + c] = sh[256]; }
}

// BN + ReLU: read fp32 conv output, emit bf16 activation for the next layer.
__global__ __launch_bounds__(256)
void bn_apply(const float* __restrict__ X, const float* __restrict__ stats,
              const float* __restrict__ nb, unsigned short* __restrict__ Ybf,
              int N, int C)
{
  size_t i = (size_t)blockIdx.x * 256 + threadIdx.x;
  if (i >= (size_t)N * C) return;
  int c = (int)(i % C);
  float m  = stats[c] / (float)N;
  float vv = stats[C + c] / (float)N - m * m;     // biased variance
  float g = nb[c], b = nb[C + c];
  float y = (X[i] - m) * (g * rsqrtf(vv + 1e-5f)) + b;
  Ybf[i] = f2bf(y > 0.f ? y : 0.f);
}

// ---------------------------------------------------------------------------

extern "C" void kernel_launch(void* const* d_in, const int* in_sizes, int n_in,
                              void* d_out, int out_size, void* d_ws, size_t ws_size,
                              hipStream_t stream) {
  (void)n_in; (void)out_size; (void)ws_size;
  static const int CS[4] = {64, 96, 160, 256};

  // ----- inputs (setup_inputs insertion order) -----
  const float* feats = (const float*)d_in[0];
  const float* P     = (const float*)d_in[1];
  const int* nbrL[4] = {(const int*)d_in[2], (const int*)d_in[3],
                        (const int*)d_in[4], (const int*)d_in[5]};
  const int* dnL[3]  = {(const int*)d_in[6], (const int*)d_in[9],  (const int*)d_in[12]};
  const int* parL[3] = {(const int*)d_in[7], (const int*)d_in[10], (const int*)d_in[13]};
  const int* koffL[3]= {(const int*)d_in[8], (const int*)d_in[11], (const int*)d_in[14]};

  const int Nl[4] = {in_sizes[2] / 27, in_sizes[3] / 27, in_sizes[4] / 27, in_sizes[5] / 27};

  // ----- params: concatenated fp32 buffer, insertion order of _make_params -----
  size_t off = 0;
  auto take = [&](size_t n) { const float* p = P + off; off += n; return p; };
  const float *stemW1 = take(27ull * 5 * 64), *stemN1 = take(128);
  const float *stemW2 = take(27ull * 64 * 64), *stemN2 = take(128);
  const float *downW[3], *downN[3], *blkW1[3], *blkN1[3], *blkW2[3], *blkN2[3];
  for (int i = 0; i < 3; ++i) {
    int ci = CS[i], co = CS[i + 1];
    downW[i] = take(8ull * ci * co);  downN[i] = take(2ull * co);
    blkW1[i] = take(27ull * co * co); blkN1[i] = take(2ull * co);
    blkW2[i] = take(27ull * co * co); blkN2[i] = take(2ull * co);
  }
  const float *upW[3], *upN[3], *upWf[3], *ubW1[3], *ubN1[3], *ubW2[3], *ubN2[3];
  for (int k = 0; k < 3; ++k) {
    int ci = CS[3 - k], co = CS[2 - k];
    upW[k]  = take(8ull * ci * co);   upN[k]  = take(2ull * co);
    upWf[k] = take(2ull * co * co);
    ubW1[k] = take(27ull * co * co);  ubN1[k] = take(2ull * co);
    ubW2[k] = take(27ull * co * co);  ubN2[k] = take(2ull * co);
  }
  const float* outW = take(64ull * 64);
  const float* memW = take(160ull * 256);
  const float* memB = take(256);

  // ----- workspace bump allocator -----
  char* wp = (char*)d_ws;
  size_t cur = 0;
  auto allocB = [&](size_t bytes) {
    cur = (cur + 255) & ~size_t(255);
    void* p = wp + cur;
    cur += bytes;
    return p;
  };
  auto allocF = [&](size_t n) { return (float*)allocB(n * 4); };
  auto allocU = [&](size_t n) { return (unsigned short*)allocB(n * 2); };

  size_t maxE = 0;
  for (int l = 0; l < 4; ++l) {
    size_t e = (size_t)Nl[l] * CS[l];
    if (e > maxE) maxE = e;
  }
  float* F     = allocF(maxE);      // fp32 conv output / BN input (ping)
  float* stats = allocF(512);
  unsigned short* featsb = allocU((size_t)Nl[0] * 5);
  unsigned short* T1b    = allocU(maxE);
  unsigned short* T2b    = allocU(maxE);
  unsigned short* S0b    = allocU((size_t)Nl[0] * 64);
  unsigned short* S1b    = allocU((size_t)Nl[1] * 96);
  unsigned short* S2b    = allocU((size_t)Nl[2] * 160);
  unsigned short* L3b    = allocU((size_t)Nl[3] * 256);

  // ----- prepack all weights to bf16 B-fragments -----
  auto packw = [&](const float* W, int K, int Cin, int Co) {
    int nC = (Cin + 31) / 32, nT = Co / 16;
    size_t total = (size_t)K * nC * nT * 512;
    unsigned short* p = allocU(total);
    pack_w<<<dim3((unsigned)((total + 255) / 256)), 256, 0, stream>>>(W, p, K, Cin, Co);
    return p;
  };
  const unsigned short* stemW1p = packw(stemW1, 27, 5, 64);
  const unsigned short* stemW2p = packw(stemW2, 27, 64, 64);
  const unsigned short *downWp[3], *blkW1p[3], *blkW2p[3];
  for (int i = 0; i < 3; ++i) {
    downWp[i] = packw(downW[i], 8, CS[i], CS[i + 1]);
    blkW1p[i] = packw(blkW1[i], 27, CS[i + 1], CS[i + 1]);
    blkW2p[i] = packw(blkW2[i], 27, CS[i + 1], CS[i + 1]);
  }
  const unsigned short *upWp[3], *upWfTp[3], *upWfBp[3], *ubW1p[3], *ubW2p[3];
  for (int k = 0; k < 3; ++k) {
    int ci = CS[3 - k], co = CS[2 - k];
    upWp[k]   = packw(upW[k], 8, ci, co);
    upWfTp[k] = packw(upWf[k], 1, co, co);
    upWfBp[k] = packw(upWf[k] + (size_t)co * co, 1, co, co);
    ubW1p[k]  = packw(ubW1[k], 27, co, co);
    ubW2p[k]  = packw(ubW2[k], 27, co, co);
  }
  const unsigned short* outWp = packw(outW, 1, 64, 64);
  const unsigned short* memWp = packw(memW, 1, 160, 256);

  // ----- launch helpers -----
  auto grid_of = [](int N, int Co) { return dim3((N + 127) / 128, (Co + 63) / 64); };
  auto conv_nbr = [&](const unsigned short* X, int Cin, const unsigned short* W,
                      const int* nbr, float* Y, unsigned short* Ybf,
                      int N, int Co, int K) {
    if ((Cin & 31) == 0)
      gconv_nbr_a<<<grid_of(N, Co), 128, 0, stream>>>(X, W, nbr, nullptr, nullptr,
                                                      nullptr, Y, Ybf, N, Cin, Co, K, 0);
    else
      gconv_nbr_u<<<grid_of(N, Co), 128, 0, stream>>>(X, W, nbr, nullptr, nullptr,
                                                      nullptr, Y, Ybf, N, Cin, Co, K, 0);
  };
  auto conv_dense = [&](const unsigned short* X, int Cin, const unsigned short* W,
                        const float* bias, float* Y, unsigned short* Ybf,
                        int N, int Co, int accum) {
    gconv_dense<<<grid_of(N, Co), 128, 0, stream>>>(X, W, nullptr, nullptr, nullptr,
                                                    bias, Y, Ybf, N, Cin, Co, 1, accum);
  };
  auto conv_up = [&](const unsigned short* X, int Cin, const unsigned short* W,
                     const int* par, const int* koff, float* Y,
                     int N, int Co) {
    gconv_up<<<grid_of(N, Co), 128, 0, stream>>>(X, W, nullptr, par, koff,
                                                 nullptr, Y, nullptr, N, Cin, Co, 8, 0);
  };
  auto bnrelu = [&](const float* Xf, int N, int C, const float* nb, unsigned short* Ybf) {
    bn_reduce<<<dim3(C), 256, 0, stream>>>(Xf, stats, N, C);
    size_t tot = (size_t)N * C;
    bn_apply<<<dim3((unsigned)((tot + 255) / 256)), 256, 0, stream>>>(Xf, stats, nb, Ybf, N, C);
  };

  float* full_out = (float*)d_out;                       // [N0, 64]
  float* mem_out  = (float*)d_out + (size_t)Nl[0] * 64;  // [N2, 256]

  // ----- feats -> bf16 -----
  {
    size_t n = (size_t)Nl[0] * 5;
    f32_to_bf16<<<dim3((unsigned)((n + 255) / 256)), 256, 0, stream>>>(feats, featsb, n);
  }

  // ----- stem -----
  conv_nbr(featsb, 5, stemW1p, nbrL[0], F, nullptr, Nl[0], 64, 27);
  bnrelu(F, Nl[0], 64, stemN1, T1b);
  conv_nbr(T1b, 64, stemW2p, nbrL[0], F, nullptr, Nl[0], 64, 27);
  bnrelu(F, Nl[0], 64, stemN2, S0b);

  // ----- encoder -----
  const unsigned short* encIn[3] = {S0b, S1b, S2b};
  unsigned short* encOut[3] = {S1b, S2b, L3b};
  for (int i = 0; i < 3; ++i) {
    int ci = CS[i], co = CS[i + 1], N = Nl[i + 1];
    conv_nbr(encIn[i], ci, downWp[i], dnL[i], F, nullptr, N, co, 8);
    bnrelu(F, N, co, downN[i], T1b);
    conv_nbr(T1b, co, blkW1p[i], nbrL[i + 1], F, nullptr, N, co, 27);
    bnrelu(F, N, co, blkN1[i], T2b);
    conv_nbr(T2b, co, blkW2p[i], nbrL[i + 1], F, nullptr, N, co, 27);
    bnrelu(F, N, co, blkN2[i], encOut[i]);
  }

  // ----- memory head: mem_feat = levels[2] @ mem_W + mem_b (fp32 out) -----
  conv_dense(S2b, 160, memWp, memB, mem_out, nullptr, Nl[2], 256, 0);

  // ----- decoder -----
  const unsigned short* skips[3] = {S2b, S1b, S0b};
  const unsigned short* curb = L3b;
  for (int k = 0; k < 3; ++k) {
    int d = 2 - k;
    int ci = CS[3 - k], co = CS[2 - k], N = Nl[d];
    // inverse conv (parent/koff rulebook)
    conv_up(curb, ci, upWp[k], parL[d], koffL[d], F, N, co);
    bnrelu(F, N, co, upN[k], T1b);
    // concat + 1x1 fuse as two GEMMs (second pass emits bf16)
    conv_dense(T1b, co, upWfTp[k], nullptr, F, nullptr, N, co, 0);
    conv_dense(skips[k], co, upWfBp[k], nullptr, F, T2b, N, co, 1);
    // submanifold block
    conv_nbr(T2b, co, ubW1p[k], nbrL[d], F, nullptr, N, co, 27);
    bnrelu(F, N, co, ubN1[k], T1b);
    conv_nbr(T1b, co, ubW2p[k], nbrL[d], F, nullptr, N, co, 27);
    bnrelu(F, N, co, ubN2[k], T2b);
    curb = T2b;
  }

  // ----- output projection (fp32 out) -----
  conv_dense(curb, 64, outWp, nullptr, full_out, nullptr, Nl[0], 64, 0);
}
